// CGFA_19533511262348
// MI455X (gfx1250) — matrix operand order repaired and checked by reading, not compiled, not run on hardware
//
#include <hip/hip_runtime.h>
#include <math.h>

// MI455X / gfx1250: wave32, WMMA 16x16x32 bf16 -> f32 accumulate.
// One workgroup (256 thr = 8 waves) per batch element; whole pipeline fused
// so only raw inputs (~406 MB) cross HBM (~17us at 23.3 TB/s).
// All WMMA B-operands are staged COLUMN-MAJOR in LDS so each lane's fragment
// is two runs of 8 contiguous bf16 -> two ds_load_b128 (no strided u16 loads / packing).

typedef __attribute__((ext_vector_type(16))) __bf16 bf16x16;
typedef __attribute__((ext_vector_type(8)))  __bf16 bf16x8;
typedef __attribute__((ext_vector_type(8)))  float  f32x8;

#define WSTRIDE 264  // column stride (elements) for staged weights, K<=256
#define CSTRIDE 72   // column stride (elements) for ax/emb col-major, K<=64

#define SMEM_BYTES 268288
// LDS carve (bytes), all offsets 16B aligned:
#define OFF_W    0        // bf16[128 cols * 264] weights col-major (Wa/Wu/Aff/Wc)
#define OFF_A1   67584    // bf16[64*64]  An1 row-major, later s_sm   (A operand)
#define OFF_A2   75776    // bf16[64*64]  An2 row-major, later sT_sm  (A operand)
#define OFF_X1   83968    // bf16[64*128] x1 -> emb1 row-major        (A / BT)
#define OFF_X2   100352   // bf16[64*128] x2 -> emb2 row-major        (A / BT)
#define OFF_XC1  116736   // bf16[128 cols * 72] ax1 col -> emb1 col  (B operand)
#define OFF_XC2  135168   // bf16[128 cols * 72] ax2 col -> emb2 col  (B operand)
#define OFF_T1   153600   // bf16[64*128] t1 -> r1 row-major          (A operand)
#define OFF_T2   169984   // bf16[64*128] r2 row-major                (A operand)
#define OFF_S    186368   // float[64*64] colsums -> raw s -> pooling scratch
#define OFF_N1   202752   // float[64*128] new1
#define OFF_N2   235520   // float[64*128] new2

static __device__ __forceinline__ f32x8 zero8() { f32x8 c = {}; return c; }

static __device__ __forceinline__ bf16x16 cat8(bf16x8 lo, bf16x8 hi) {
  return __builtin_shufflevector(lo, hi, 0, 1, 2, 3, 4, 5, 6, 7,
                                 8, 9, 10, 11, 12, 13, 14, 15);
}

static __device__ __forceinline__ f32x8 mma_bf16(bf16x16 a, bf16x16 b, f32x8 c) {
  // (neg_a, A, neg_b, B, c_mod, C, reuse_a, reuse_b)
  return __builtin_amdgcn_wmma_f32_16x16x32_bf16(false, a, false, b, (short)0, c,
                                                 false, false);
}

// A fragment: 16x32 (MxK) tile from row-major LDS, leading dim ld.
// lanes 0-15: M=lane, K={0..7,16..23}; lanes 16-31: M=lane-16, K={8..15,24..31}.
// Both K groups are 16B-contiguous per lane.
static __device__ __forceinline__ bf16x16 ldA(const __bf16* s, int ld, int r0, int c0) {
  const int lane = threadIdx.x & 31;
  const __bf16* p = s + (r0 + (lane & 15)) * ld + c0 + ((lane >> 4) << 3);
  return cat8(*(const bf16x8*)p, *(const bf16x8*)(p + 16));
}

// B fragment: 32x16 (KxN) tile from COLUMN-MAJOR LDS (col stride S elements):
// element (k,n) at s[(n)*S + k].  lanes 0-15: N=lane, K={0..7,16..23};
// lanes 16-31: N=lane-16, K={8..15,24..31} -> two 16B chunks per lane.
// Also serves as "B = rowmajor^T" loader (pass S = row stride).
static __device__ __forceinline__ bf16x16 ldBc(const __bf16* s, int S, int k0, int n0) {
  const int lane = threadIdx.x & 31;
  const __bf16* p = s + (n0 + (lane & 15)) * S + k0 + ((lane >> 4) << 3);
  return cat8(*(const bf16x8*)p, *(const bf16x8*)(p + 16));
}

// C/D 16x16 f32: lane gives N=lane&15, rows = vgpr + 8*(lane>=16).
static __device__ __forceinline__ void stC(float* d, int ld, int r0, int c0, f32x8 c) {
  const int lane = threadIdx.x & 31;
  float* p = d + (r0 + ((lane >> 4) << 3)) * ld + c0 + (lane & 15);
#pragma unroll
  for (int r = 0; r < 8; ++r) p[r * ld] = c[r];
}

static __device__ __forceinline__ void stCbf(__bf16* d, int ld, int r0, int c0, f32x8 c) {
  const int lane = threadIdx.x & 31;
  __bf16* p = d + (r0 + ((lane >> 4) << 3)) * ld + c0 + (lane & 15);
#pragma unroll
  for (int r = 0; r < 8; ++r) p[r * ld] = (__bf16)c[r];
}

// Store C tile transposed into a column-major buffer (col stride S):
// lane holds 8 consecutive rows of one column -> single 16B contiguous store.
static __device__ __forceinline__ void stCcol(__bf16* d, int S, int r0, int c0, f32x8 c) {
  const int lane = threadIdx.x & 31;
  __bf16* p = d + (c0 + (lane & 15)) * S + r0 + ((lane >> 4) << 3);
  bf16x8 v;
#pragma unroll
  for (int r = 0; r < 8; ++r) v[r] = (__bf16)c[r];
  *(bf16x8*)p = v;
}

static __device__ __forceinline__ f32x8 relu8(f32x8 c) {
#pragma unroll
  for (int r = 0; r < 8; ++r) c[r] = fmaxf(c[r], 0.0f);
  return c;
}

static __device__ __forceinline__ f32x8 bias_row(const float* bias, int c0) {
  const float v = bias[c0 + (threadIdx.x & 15)];
  f32x8 c;
#pragma unroll
  for (int r = 0; r < 8; ++r) c[r] = v;
  return c;
}

// Stage a row-major [K x 128] fp32 weight into column-major bf16 LDS.
static __device__ __forceinline__ void cvt_copy_T(const float* __restrict__ g,
                                                  __bf16* __restrict__ s, int K) {
  for (int i = threadIdx.x; i < K * 128; i += 256) {
    const int k = i >> 7, n = i & 127;
    s[n * WSTRIDE + k] = (__bf16)g[i];
  }
}

__global__ __launch_bounds__(256, 1) void gmn_fwd(
    const float* __restrict__ A_src, const float* __restrict__ X_src,
    const unsigned char* __restrict__ M_src,
    const float* __restrict__ A_dst, const float* __restrict__ X_dst,
    const unsigned char* __restrict__ M_dst,
    const float* __restrict__ Wa, const float* __restrict__ ba,
    const float* __restrict__ Wu, const float* __restrict__ bu,
    const float* __restrict__ Aff,
    const float* __restrict__ Wc, const float* __restrict__ bc,
    const float* __restrict__ Wp1, const float* __restrict__ Wp2,
    float* __restrict__ out, int Bn) {
  extern __shared__ char smem[];
  __bf16* sW   = (__bf16*)(smem + OFF_W);
  __bf16* sA1  = (__bf16*)(smem + OFF_A1);
  __bf16* sA2  = (__bf16*)(smem + OFF_A2);
  __bf16* sX1  = (__bf16*)(smem + OFF_X1);
  __bf16* sX2  = (__bf16*)(smem + OFF_X2);
  __bf16* sXc1 = (__bf16*)(smem + OFF_XC1);
  __bf16* sXc2 = (__bf16*)(smem + OFF_XC2);
  __bf16* sT1  = (__bf16*)(smem + OFF_T1);
  __bf16* sT2  = (__bf16*)(smem + OFF_T2);
  float*  sS   = (float*)(smem + OFF_S);
  float*  sN1  = (float*)(smem + OFF_N1);
  float*  sN2  = (float*)(smem + OFF_N2);

  const int tid = threadIdx.x;
  const int wv  = tid >> 5;  // wave id 0..7, owns output column strip wv*16
  const int b   = blockIdx.x;

  const float* A1 = A_src + (size_t)b * 64 * 64;
  const float* A2 = A_dst + (size_t)b * 64 * 64;
  const float* x1 = X_src + (size_t)b * 64 * 128;
  const float* x2 = X_dst + (size_t)b * 64 * 128;
  const unsigned char* m1 = M_src + (size_t)b * 64;
  const unsigned char* m2 = M_dst + (size_t)b * 64;

  // ---- stage 0: column sums of A, stage x as bf16, stage Wa ------------
  if (tid < 64) {
    float s = 0.f;
    for (int i = 0; i < 64; ++i) s += A1[i * 64 + tid];
    sS[tid] = fmaxf(s, 1e-12f);
  } else if (tid < 128) {
    const int j = tid - 64;
    float s = 0.f;
    for (int i = 0; i < 64; ++i) s += A2[i * 64 + j];
    sS[64 + j] = fmaxf(s, 1e-12f);
  }
  for (int e = tid; e < 64 * 128; e += 256) {
    sX1[e] = (__bf16)x1[e];
    sX2[e] = (__bf16)x2[e];
  }
  cvt_copy_T(Wa, sW, 128);
  __syncthreads();
  for (int e = tid; e < 64 * 64; e += 256) {
    const int j = e & 63;
    sA1[e] = (__bf16)(A1[e] / sS[j]);
    sA2[e] = (__bf16)(A2[e] / sS[64 + j]);
  }
  __syncthreads();

  // ---- stage 1: ax = relu(x @ Wa + ba) -> sXc1/sXc2 (column-major) -----
  for (int mt = 0; mt < 4; ++mt) {
    f32x8 acc = bias_row(ba, wv * 16);
#pragma unroll
    for (int kk = 0; kk < 128; kk += 32)
      acc = mma_bf16(ldA(sX1, 128, mt * 16, kk), ldBc(sW, WSTRIDE, kk, wv * 16), acc);
    stCcol(sXc1, CSTRIDE, mt * 16, wv * 16, relu8(acc));

    acc = bias_row(ba, wv * 16);
#pragma unroll
    for (int kk = 0; kk < 128; kk += 32)
      acc = mma_bf16(ldA(sX2, 128, mt * 16, kk), ldBc(sW, WSTRIDE, kk, wv * 16), acc);
    stCcol(sXc2, CSTRIDE, mt * 16, wv * 16, relu8(acc));
  }
  __syncthreads();
  cvt_copy_T(Wu, sW, 128);
  __syncthreads();

  // ---- stage 2: emb = An @ ax + relu(x @ Wu + bu) (held in regs) -------
  f32x8 e1[4], e2[4];
  for (int mt = 0; mt < 4; ++mt) {
    f32x8 acc = bias_row(bu, wv * 16);
#pragma unroll
    for (int kk = 0; kk < 128; kk += 32)
      acc = mma_bf16(ldA(sX1, 128, mt * 16, kk), ldBc(sW, WSTRIDE, kk, wv * 16), acc);
    acc = relu8(acc);
#pragma unroll
    for (int kk = 0; kk < 64; kk += 32)
      acc = mma_bf16(ldA(sA1, 64, mt * 16, kk), ldBc(sXc1, CSTRIDE, kk, wv * 16), acc);
    e1[mt] = acc;

    acc = bias_row(bu, wv * 16);
#pragma unroll
    for (int kk = 0; kk < 128; kk += 32)
      acc = mma_bf16(ldA(sX2, 128, mt * 16, kk), ldBc(sW, WSTRIDE, kk, wv * 16), acc);
    acc = relu8(acc);
#pragma unroll
    for (int kk = 0; kk < 64; kk += 32)
      acc = mma_bf16(ldA(sA2, 64, mt * 16, kk), ldBc(sXc2, CSTRIDE, kk, wv * 16), acc);
    e2[mt] = acc;
  }
  __syncthreads();  // everyone done reading x / ax before overwrite
  for (int mt = 0; mt < 4; ++mt) {
    stCbf(sX1, 128, mt * 16, wv * 16, e1[mt]);   // emb row-major (A / BT uses)
    stCbf(sX2, 128, mt * 16, wv * 16, e2[mt]);
    stCcol(sXc1, CSTRIDE, mt * 16, wv * 16, e1[mt]);  // emb col-major (B use)
    stCcol(sXc2, CSTRIDE, mt * 16, wv * 16, e2[mt]);
  }
  cvt_copy_T(Aff, sW, 128);
  __syncthreads();

  // ---- stage 3: t1 = emb1 @ Aff -> sT1 (row-major, used as A next) -----
  for (int mt = 0; mt < 4; ++mt) {
    f32x8 acc = zero8();
#pragma unroll
    for (int kk = 0; kk < 128; kk += 32)
      acc = mma_bf16(ldA(sX1, 128, mt * 16, kk), ldBc(sW, WSTRIDE, kk, wv * 16), acc);
    stCbf(sT1, 128, mt * 16, wv * 16, acc);
  }
  __syncthreads();

  // ---- stage 4: s = t1 @ emb2^T -> sS (f32); prefetch Wc into sW -------
  cvt_copy_T(Wc, sW, 256);
  for (int tt = 0; tt < 2; ++tt) {
    const int t  = wv + tt * 8;
    const int mt = t >> 2, nt = t & 3;
    f32x8 acc = zero8();
#pragma unroll
    for (int kk = 0; kk < 128; kk += 32)  // B = emb2^T: row-major emb2 is K-contig
      acc = mma_bf16(ldA(sT1, 128, mt * 16, kk), ldBc(sX2, 128, kk, nt * 16), acc);
    stC(sS, 64, mt * 16, nt * 16, acc);
  }
  __syncthreads();

  // ---- stage 5: masked row softmax -> sA1, masked col softmax -> sA2 ---
  if (tid < 64) {
    const int n = tid;
    float mx = -1e30f;
    for (int m = 0; m < 64; ++m)
      mx = fmaxf(mx, m2[m] ? sS[n * 64 + m] : -1e9f);
    float sum = 0.f;
    for (int m = 0; m < 64; ++m)
      sum += expf((m2[m] ? sS[n * 64 + m] : -1e9f) - mx);
    const float inv = 1.0f / sum;
    for (int m = 0; m < 64; ++m)
      sA1[n * 64 + m] = (__bf16)(expf((m2[m] ? sS[n * 64 + m] : -1e9f) - mx) * inv);
  } else if (tid < 128) {
    const int mcol = tid - 64;
    float mx = -1e30f;
    for (int n = 0; n < 64; ++n)
      mx = fmaxf(mx, m1[n] ? sS[n * 64 + mcol] : -1e9f);
    float sum = 0.f;
    for (int n = 0; n < 64; ++n)
      sum += expf((m1[n] ? sS[n * 64 + mcol] : -1e9f) - mx);
    const float inv = 1.0f / sum;
    for (int n = 0; n < 64; ++n)
      sA2[mcol * 64 + n] = (__bf16)(expf((m1[n] ? sS[n * 64 + mcol] : -1e9f) - mx) * inv);
  }
  __syncthreads();

  // ---- stage 6: r1 = s_sm @ emb2 -> sT1 ; r2 = sT_sm @ emb1 -> sT2 -----
  for (int mt = 0; mt < 4; ++mt) {
    f32x8 acc = zero8();
#pragma unroll
    for (int kk = 0; kk < 64; kk += 32)
      acc = mma_bf16(ldA(sA1, 64, mt * 16, kk), ldBc(sXc2, CSTRIDE, kk, wv * 16), acc);
    stCbf(sT1, 128, mt * 16, wv * 16, acc);

    acc = zero8();
#pragma unroll
    for (int kk = 0; kk < 64; kk += 32)
      acc = mma_bf16(ldA(sA2, 64, mt * 16, kk), ldBc(sXc1, CSTRIDE, kk, wv * 16), acc);
    stCbf(sT2, 128, mt * 16, wv * 16, acc);
  }
  __syncthreads();

  // ---- stage 7: new = concat(emb, r) @ Wc + bc -> sN (f32) -------------
  for (int mt = 0; mt < 4; ++mt) {
    f32x8 acc = bias_row(bc, wv * 16);
#pragma unroll
    for (int kk = 0; kk < 128; kk += 32)
      acc = mma_bf16(ldA(sX1, 128, mt * 16, kk), ldBc(sW, WSTRIDE, kk, wv * 16), acc);
#pragma unroll
    for (int kk = 0; kk < 128; kk += 32)
      acc = mma_bf16(ldA(sT1, 128, mt * 16, kk), ldBc(sW, WSTRIDE, 128 + kk, wv * 16), acc);
    stC(sN1, 128, mt * 16, wv * 16, acc);

    acc = bias_row(bc, wv * 16);
#pragma unroll
    for (int kk = 0; kk < 128; kk += 32)
      acc = mma_bf16(ldA(sX2, 128, mt * 16, kk), ldBc(sW, WSTRIDE, kk, wv * 16), acc);
#pragma unroll
    for (int kk = 0; kk < 128; kk += 32)
      acc = mma_bf16(ldA(sT2, 128, mt * 16, kk), ldBc(sW, WSTRIDE, 128 + kk, wv * 16), acc);
    stC(sN2, 128, mt * 16, wv * 16, acc);
  }
  __syncthreads();

  // ---- stage 8: attention pooling (threads 0-127: graph1, 128-255: g2) -
  const int gsel = tid >> 7;
  const int u    = tid & 127;
  const float* sN = gsel ? sN2 : sN1;
  const unsigned char* mg = gsel ? m2 : m1;
  const float* Wp = gsel ? Wp2 : Wp1;
  float* red = sS + gsel * 512;  // [0,128) mean, [128,256) ctx, [256,320) scores
  float* og = out + (size_t)gsel * Bn * 128 + (size_t)b * 128;

  {  // masked mean
    float acc = 0.f, num = 0.f;
    for (int n = 0; n < 64; ++n) {
      const float mv = mg[n] ? 1.0f : 0.0f;
      acc += mv * sN[n * 128 + u];
      num += mv;
    }
    red[u] = acc / fmaxf(num, 1.0f);
  }
  __syncthreads();
  {  // ctx = tanh(mean @ Wp)
    float acc = 0.f;
    for (int d = 0; d < 128; ++d) acc += red[d] * Wp[d * 128 + u];
    red[128 + u] = tanhf(acc);
  }
  __syncthreads();
  if (u < 64) {  // scores = sigmoid(xm . ctx)
    float acc = 0.f;
    for (int d = 0; d < 128; ++d) acc += sN[u * 128 + d] * red[128 + d];
    acc = mg[u] ? acc : 0.0f;
    red[256 + u] = 1.0f / (1.0f + expf(-acc));
  }
  __syncthreads();
  {  // g = sum_n xm[n,:] * score[n]
    float acc = 0.f;
    for (int n = 0; n < 64; ++n) {
      const float mv = mg[n] ? 1.0f : 0.0f;
      acc += mv * sN[n * 128 + u] * red[256 + n];
    }
    og[u] = acc;
  }
}

extern "C" void kernel_launch(void* const* d_in, const int* in_sizes, int n_in,
                              void* d_out, int out_size, void* d_ws, size_t ws_size,
                              hipStream_t stream) {
  (void)n_in; (void)out_size; (void)d_ws; (void)ws_size;
  const float* A_src = (const float*)d_in[0];
  const float* X_src = (const float*)d_in[1];
  const unsigned char* M_src = (const unsigned char*)d_in[2];
  const float* A_dst = (const float*)d_in[3];
  const float* X_dst = (const float*)d_in[4];
  const unsigned char* M_dst = (const unsigned char*)d_in[5];
  const float* Wa  = (const float*)d_in[6];
  const float* ba  = (const float*)d_in[7];
  const float* Wu  = (const float*)d_in[8];
  const float* bu  = (const float*)d_in[9];
  const float* Aff = (const float*)d_in[10];
  const float* Wc  = (const float*)d_in[11];
  const float* bc  = (const float*)d_in[12];
  const float* Wp1 = (const float*)d_in[13];
  const float* Wp2 = (const float*)d_in[14];
  const int Bn = in_sizes[0] / (64 * 64);

  hipFuncSetAttribute(reinterpret_cast<const void*>(gmn_fwd),
                      hipFuncAttributeMaxDynamicSharedMemorySize, SMEM_BYTES);
  gmn_fwd<<<Bn, 256, SMEM_BYTES, stream>>>(A_src, X_src, M_src, A_dst, X_dst, M_dst,
                                           Wa, ba, Wu, bu, Aff, Wc, bc, Wp1, Wp2,
                                           (float*)d_out, Bn);
}
